// UltrametricCausalSelfAttention_37245956391233
// MI455X (gfx1250) — matrix operand order repaired and checked by reading, not compiled, not run on hardware
//
#include <hip/hip_runtime.h>

#define B_  2
#define T_  1024
#define C_  1024
#define H_  16
#define HD_ 64
#define KD_ 8

typedef __bf16 bf16_t;
typedef __bf16 v16bf __attribute__((ext_vector_type(16)));
typedef float  v8f   __attribute__((ext_vector_type(8)));
typedef unsigned int v4u __attribute__((ext_vector_type(4)));

union Frag32 { v4u u[2]; v16bf v; };

// ---------------------------------------------------------------------------
// f32 -> bf16 elementwise convert
// ---------------------------------------------------------------------------
__global__ void cvt_bf16(const float* __restrict__ in, bf16_t* __restrict__ out, int n) {
  int i = blockIdx.x * blockDim.x + threadIdx.x;
  if (i < n) out[i] = (bf16_t)in[i];
}

// ---------------------------------------------------------------------------
// W[k][n] f32  ->  Wt[n][k] bf16   (1024x1024), LDS-tiled transpose
// ---------------------------------------------------------------------------
__global__ void transpose_cvt(const float* __restrict__ W, bf16_t* __restrict__ Wt) {
  __shared__ float tile[32][33];
  const int n0 = blockIdx.x * 32;
  const int k0 = blockIdx.y * 32;
  const int tx = threadIdx.x;           // 0..31
  for (int r = threadIdx.y; r < 32; r += 8)
    tile[r][tx] = W[(size_t)(k0 + r) * 1024 + n0 + tx];
  __syncthreads();
  for (int r = threadIdx.y; r < 32; r += 8)
    Wt[(size_t)(n0 + r) * 1024 + k0 + tx] = (bf16_t)tile[tx][r];
}

// ---------------------------------------------------------------------------
// C[M][N] f32 = A[M][Kd] bf16 @ Bt[N][Kd]^T bf16   (TN, K-contiguous both sides)
// block = 128 threads (4 waves); wave tile = 32(M) x 64(N)
// ---------------------------------------------------------------------------
__global__ __launch_bounds__(128) void gemm_bf16_tn(
    const bf16_t* __restrict__ A, const bf16_t* __restrict__ Bt,
    float* __restrict__ Cm, int M, int N, int Kd)
{
  const int lane = threadIdx.x & 31;
  const int wv   = threadIdx.x >> 5;
  const int li   = lane & 15;
  const int half = lane >> 4;
  const int kb   = half * 8;
  const int row0 = blockIdx.x * 32;
  const int col0 = blockIdx.y * 256 + wv * 64;

  v8f acc[2][4] = {};

  const bf16_t* a0p = A + (size_t)(row0 + li) * Kd;
  const bf16_t* a1p = a0p + (size_t)16 * Kd;

  for (int k0 = 0; k0 < Kd; k0 += 32) {
    Frag32 a0, a1;
    a0.u[0] = *(const v4u*)(a0p + k0 + kb);
    a0.u[1] = *(const v4u*)(a0p + k0 + kb + 16);
    a1.u[0] = *(const v4u*)(a1p + k0 + kb);
    a1.u[1] = *(const v4u*)(a1p + k0 + kb + 16);
#pragma unroll
    for (int nf = 0; nf < 4; ++nf) {
      Frag32 b;
      const bf16_t* bp = Bt + (size_t)(col0 + nf * 16 + li) * Kd + k0 + half * 8;
      b.u[0] = *(const v4u*)(bp);
      b.u[1] = *(const v4u*)(bp + 16);
      acc[0][nf] = __builtin_amdgcn_wmma_f32_16x16x32_bf16(
          false, a0.v, false, b.v, (short)0, acc[0][nf], false, false);
      acc[1][nf] = __builtin_amdgcn_wmma_f32_16x16x32_bf16(
          false, a1.v, false, b.v, (short)0, acc[1][nf], false, false);
    }
  }
#pragma unroll
  for (int mi = 0; mi < 2; ++mi)
#pragma unroll
    for (int nf = 0; nf < 4; ++nf) {
      float* cp = Cm + (size_t)(row0 + mi * 16 + 8 * half) * N + col0 + nf * 16 + li;
#pragma unroll
      for (int r = 0; r < 8; ++r) cp[(size_t)r * N] = acc[mi][nf][r];
    }
}

// ---------------------------------------------------------------------------
// RoPE + RMSNorm + qd/kd = 3*sigmoid(row @ Wd).  One wave per (b,t,h) row.
// qk: [B*T][H*HD] f32 ; cos/sin: [T][32] ; Wd: [64][8] ; out: [(b*H+h)*T+t][8]
// ---------------------------------------------------------------------------
__global__ __launch_bounds__(256) void rope_rms_dirs(
    const float* __restrict__ qk, const float* __restrict__ cosb,
    const float* __restrict__ sinb, const float* __restrict__ Wd,
    float* __restrict__ out)
{
  const int wid  = (blockIdx.x * blockDim.x + threadIdx.x) >> 5;
  const int lane = threadIdx.x & 31;
  if (wid >= B_ * T_ * H_) return;
  const int h = wid % H_;
  const int t = (wid / H_) % T_;
  const int b = wid / (H_ * T_);

  const float* row = qk + (size_t)(b * T_ + t) * (H_ * HD_) + h * HD_;
  float x1 = row[lane];
  float x2 = row[lane + 32];
  float c  = cosb[t * 32 + lane];
  float s  = sinb[t * 32 + lane];
  float o1 =  x1 * c + x2 * s;
  float o2 = -x1 * s + x2 * c;

  float ss = o1 * o1 + o2 * o2;
#pragma unroll
  for (int off = 16; off; off >>= 1) ss += __shfl_xor(ss, off);
  float inv = rsqrtf(ss * (1.0f / 64.0f) + 1.1920929e-7f);
  o1 *= inv; o2 *= inv;

  float acc[KD_];
#pragma unroll
  for (int d = 0; d < KD_; ++d)
    acc[d] = o1 * Wd[lane * KD_ + d] + o2 * Wd[(lane + 32) * KD_ + d];
#pragma unroll
  for (int off = 16; off; off >>= 1)
#pragma unroll
    for (int d = 0; d < KD_; ++d) acc[d] += __shfl_xor(acc[d], off);

  if (lane == 0) {
    float* o = out + (size_t)((b * H_ + h) * T_ + t) * KD_;
#pragma unroll
    for (int d = 0; d < KD_; ++d) o[d] = 3.0f / (1.0f + __expf(-acc[d]));
  }
}

// ---------------------------------------------------------------------------
// v[(b*T+t)][h*64+e] f32  ->  vT[b][h][e][t] bf16  (B-fragment friendly layout)
// ---------------------------------------------------------------------------
__global__ void vprep(const float* __restrict__ v, bf16_t* __restrict__ vT) {
  int idx = blockIdx.x * blockDim.x + threadIdx.x;     // t fastest
  if (idx >= B_ * H_ * HD_ * T_) return;
  int t = idx & (T_ - 1);
  int e = (idx >> 10) & (HD_ - 1);
  int h = (idx >> 16) & (H_ - 1);
  int b = idx >> 20;
  vT[idx] = (bf16_t)v[(size_t)(b * T_ + t) * (H_ * HD_) + h * HD_ + e];
}

// ---------------------------------------------------------------------------
// Fused ultrametric scores + causal mask + row-normalized attn @ v.
// grid = (B*H, T/64), block = 128 (4 waves; wave = 16 query rows).
// Scores built directly in WMMA A-fragment layout; v consumed via vT.
// ---------------------------------------------------------------------------
__global__ __launch_bounds__(128) void ultra_attn(
    const float* __restrict__ qd, const float* __restrict__ kd,
    const bf16_t* __restrict__ vT, bf16_t* __restrict__ outb)
{
  __shared__ float kds[T_ * KD_];          // 32 KB: whole kd for this (b,h)
  const int bh   = blockIdx.x;
  const int b    = bh / H_;
  const int h    = bh % H_;
  const int wv   = threadIdx.x >> 5;
  const int lane = threadIdx.x & 31;
  const int li   = lane & 15;
  const int half = lane >> 4;
  const int kb   = half * 8;

  const float* kdg = kd + (size_t)bh * T_ * KD_;
  for (int i = threadIdx.x; i < T_ * KD_; i += 128) kds[i] = kdg[i];
  __syncthreads();

  const int irow = blockIdx.y * 64 + wv * 16;
  const int i    = irow + li;              // query row owned by this lane

  float qdv[KD_];
  const float* qrow = qd + (size_t)(bh * T_ + i) * KD_;
  float4 q0 = *(const float4*)qrow;
  float4 q1 = *(const float4*)(qrow + 4);
  qdv[0]=q0.x; qdv[1]=q0.y; qdv[2]=q0.z; qdv[3]=q0.w;
  qdv[4]=q1.x; qdv[5]=q1.y; qdv[6]=q1.z; qdv[7]=q1.w;

  v8f acc[4] = {};
  float rowsum = 0.0f;
  const bf16_t* vbase = vT + (size_t)bh * HD_ * T_;
  const int jend = irow + 16;

  for (int j0 = 0; j0 < jend; j0 += 32) {
    Frag32 af;
    const bool full = (j0 + 31 <= irow);   // whole 32-wide tile is causal
#pragma unroll
    for (int e = 0; e < 16; ++e) {
      const int j = j0 + kb + e + ((e >> 3) << 3);   // A-frag K mapping
      const float* kr = &kds[j * KD_];
      float4 k0v = *(const float4*)kr;
      float4 k1v = *(const float4*)(kr + 4);
      float kk[KD_] = {k0v.x,k0v.y,k0v.z,k0v.w,k1v.x,k1v.y,k1v.z,k1v.w};
      float csum = 0.0f, lcp = 0.0f;
#pragma unroll
      for (int d = 0; d < KD_; ++d) {
        float diff = qdv[d] - kk[d];
        csum = fmaf(-4.0f * diff, diff, csum);  // -beta * diff^2, cumulative
        lcp += __expf(csum);                    // prefix product via exp(cumsum)
      }
      float w = exp2f(lcp);                     // LOG_ALPHA == ln(2)
      if (!full && j > i) w = 0.0f;
      rowsum += w;
      af.v[e] = (bf16_t)w;
    }
#pragma unroll
    for (int nf = 0; nf < 4; ++nf) {
      Frag32 bf_;
      const bf16_t* bp = vbase + (size_t)(nf * 16 + li) * T_ + j0 + half * 8;
      bf_.u[0] = *(const v4u*)(bp);
      bf_.u[1] = *(const v4u*)(bp + 16);
      acc[nf] = __builtin_amdgcn_wmma_f32_16x16x32_bf16(
          false, af.v, false, bf_.v, (short)0, acc[nf], false, false);
    }
  }

  rowsum += __shfl_xor(rowsum, 16);        // lane li and li+16 both hold row li sum
  float inv[8];
#pragma unroll
  for (int r = 0; r < 8; ++r) {
    float rs = __shfl(rowsum, r + 8 * half);      // row for D VGPR r
    inv[r] = 1.0f / fmaxf(rs, 1e-9f);
  }
#pragma unroll
  for (int nf = 0; nf < 4; ++nf)
#pragma unroll
    for (int r = 0; r < 8; ++r) {
      const int t = irow + r + 8 * half;
      outb[(size_t)(b * T_ + t) * (H_ * HD_) + h * HD_ + nf * 16 + li] =
          (bf16_t)(acc[nf][r] * inv[r]);
    }
}

// ---------------------------------------------------------------------------
extern "C" void kernel_launch(void* const* d_in, const int* in_sizes, int n_in,
                              void* d_out, int out_size, void* d_ws, size_t ws_size,
                              hipStream_t stream) {
  (void)in_sizes; (void)n_in; (void)out_size; (void)ws_size;
  const float* x     = (const float*)d_in[0];
  const float* cosb  = (const float*)d_in[1];
  const float* sinb  = (const float*)d_in[2];
  const float* Wq    = (const float*)d_in[3];
  const float* Wk    = (const float*)d_in[4];
  const float* Wv    = (const float*)d_in[5];
  const float* Wproj = (const float*)d_in[6];
  const float* Wdq   = (const float*)d_in[7];
  const float* Wdk   = (const float*)d_in[8];
  float* y = (float*)d_out;

  char* ws = (char*)d_ws;
  const size_t MB = 1ull << 20;
  bf16_t* xbf   = (bf16_t*)(ws + 0);        // 4 MB
  bf16_t* wqt   = (bf16_t*)(ws + 4  * MB);  // 2 MB
  bf16_t* wkt   = (bf16_t*)(ws + 6  * MB);  // 2 MB
  bf16_t* wvt   = (bf16_t*)(ws + 8  * MB);  // 2 MB
  bf16_t* wpt   = (bf16_t*)(ws + 10 * MB);  // 2 MB
  float*  qf    = (float*) (ws + 12 * MB);  // 8 MB
  float*  kf    = (float*) (ws + 20 * MB);  // 8 MB
  float*  vf    = (float*) (ws + 28 * MB);  // 8 MB
  float*  qdbuf = (float*) (ws + 36 * MB);  // 1 MB
  float*  kdbuf = (float*) (ws + 37 * MB);  // 1 MB
  bf16_t* vtb   = (bf16_t*)(ws + 38 * MB);  // 4 MB
  bf16_t* outb  = (bf16_t*)(ws + 42 * MB);  // 4 MB (total 46 MB)

  cvt_bf16<<<(B_ * T_ * C_) / 256, 256, 0, stream>>>(x, xbf, B_ * T_ * C_);

  dim3 tb(32, 8), tg(32, 32);
  transpose_cvt<<<tg, tb, 0, stream>>>(Wq, wqt);
  transpose_cvt<<<tg, tb, 0, stream>>>(Wk, wkt);
  transpose_cvt<<<tg, tb, 0, stream>>>(Wv, wvt);
  transpose_cvt<<<tg, tb, 0, stream>>>(Wproj, wpt);

  dim3 gg(2048 / 32, 1024 / 256);
  gemm_bf16_tn<<<gg, 128, 0, stream>>>(xbf, wqt, qf, 2048, 1024, 1024);
  gemm_bf16_tn<<<gg, 128, 0, stream>>>(xbf, wkt, kf, 2048, 1024, 1024);
  gemm_bf16_tn<<<gg, 128, 0, stream>>>(xbf, wvt, vf, 2048, 1024, 1024);

  rope_rms_dirs<<<(B_ * T_ * H_ * 32) / 256, 256, 0, stream>>>(qf, cosb, sinb, Wdq, qdbuf);
  rope_rms_dirs<<<(B_ * T_ * H_ * 32) / 256, 256, 0, stream>>>(kf, cosb, sinb, Wdk, kdbuf);

  vprep<<<(B_ * H_ * HD_ * T_) / 256, 256, 0, stream>>>(vf, vtb);

  ultra_attn<<<dim3(B_ * H_, T_ / 64), 128, 0, stream>>>(qdbuf, kdbuf, vtb, outb);

  gemm_bf16_tn<<<gg, 128, 0, stream>>>(outb, wpt, y, 2048, 1024, 1024);
}
